// Attention_Net_40381282517811
// MI455X (gfx1250) — matrix operand order repaired and checked by reading, not compiled, not run on hardware
//
#include <hip/hip_runtime.h>
#include <math.h>

// ---------------------------------------------------------------------------
// NTS-Net style attention network for MI455X (gfx1250, wave32, WMMA).
// Convs = implicit GEMM on v_wmma_f32_16x16x32_bf16, NHWC activations,
// weights prepacked [Cout][KH*KW*Cin]. Wave tile = 32x32 (2 M-subtiles x
// 2 N-tiles -> 4 wmma/chunk reusing both A and B fragments). A staged in
// double-buffered LDS (1 barrier per K-chunk); B = 2x global_load_b128/lane.
// ---------------------------------------------------------------------------

typedef __attribute__((ext_vector_type(16))) __bf16 bf16x16;
typedef __attribute__((ext_vector_type(8)))  __bf16 bf16x8;
typedef __attribute__((ext_vector_type(8)))  float  f32x8;

#define N_ANCH 2793

// ------------------------------ small helpers ------------------------------

__global__ void foldbn_kernel(const float* __restrict__ g, const float* __restrict__ b,
                              const float* __restrict__ m, const float* __restrict__ v,
                              float* __restrict__ scale, float* __restrict__ bias, int C) {
  int i = blockIdx.x * blockDim.x + threadIdx.x;
  if (i < C) {
    float s = g[i] * rsqrtf(v[i] + 1e-5f);
    scale[i] = s;
    bias[i]  = b[i] - m[i] * s;
  }
}

// OIHW f32 -> [Cout][(kh*KW+kw)*Cin + ci] bf16
__global__ void __launch_bounds__(256) repack_w_kernel(const float* __restrict__ w,
                                                       __bf16* __restrict__ out,
                                                       int Cout, int Cin, int KH, int KW) {
  int idx = blockIdx.x * blockDim.x + threadIdx.x;
  int total = Cout * Cin * KH * KW;
  if (idx >= total) return;
  int ci = idx % Cin; int r = idx / Cin;
  int kw = r % KW;    r /= KW;
  int kh = r % KH;    int co = r / KH;
  out[idx] = (__bf16)w[(((size_t)co * Cin + ci) * KH + kh) * KW + kw];
}

// NCHW f32 -> NHWC bf16
__global__ void __launch_bounds__(256) nchw_to_nhwc_kernel(const float* __restrict__ in,
                                                           __bf16* __restrict__ out,
                                                           int N, int C, int H, int W) {
  int idx = blockIdx.x * blockDim.x + threadIdx.x;
  int total = N * C * H * W;
  if (idx >= total) return;
  int c = idx % C; int r = idx / C;
  int w = r % W;   r /= W;
  int h = r % H;   int n = r / H;
  out[idx] = (__bf16)in[(((size_t)n * C + c) * H + h) * W + w];
}

// ---------------- fast conv: Cin % 32 == 0, NHWC, 32x32 tile/wave ----------

__global__ void __launch_bounds__(256) conv_wmma_fast_kernel(
    const __bf16* __restrict__ in,   // NHWC [B][H][W][Cin]
    const __bf16* __restrict__ wpk,  // [Cout][KH*KW*Cin]
    const float* __restrict__ scale, const float* __restrict__ bias,
    const __bf16* __restrict__ resid,     // NHWC, same shape as output
    __bf16* __restrict__ outb, float* __restrict__ outf,
    int B, int Cin, int H, int W, int Cout, int Ho, int Wo,
    int KH, int KW, int stride, int pad, int relu,
    int obstride, int ooff)
{
  const int HoWo = Ho * Wo;
  const int Ncol = B * HoWo;
  const int Ktot = Cin * KH * KW;
  const int lane = threadIdx.x & 31;
  const int wave = threadIdx.x >> 5;
  const int m0 = blockIdx.y * 32;                // 2 M-subtiles of 16
  const int n0 = (blockIdx.x * 8 + wave) * 32;   // 2 N-tiles of 16
  const int mrow = lane & 15;
  const int hw2 = lane >> 4;

  __shared__ __bf16 As[2][1024];   // double-buffered, per-lane fragment order

  f32x8 acc[2][2];
#pragma unroll
  for (int s = 0; s < 2; s++)
#pragma unroll
    for (int t = 0; t < 2; t++)
      acc[s][t] = (f32x8){0.f,0.f,0.f,0.f,0.f,0.f,0.f,0.f};

  int hb[2], wb[2];
  size_t ibase[2];
#pragma unroll
  for (int t = 0; t < 2; t++) {
    int ncol = n0 + t * 16 + mrow;
    if (ncol >= Ncol) ncol = Ncol - 1;
    int ni = ncol / HoWo; int rc = ncol - ni * HoWo;
    int ho = rc / Wo;     int wo = rc - ho * Wo;
    hb[t] = ho * stride - pad;
    wb[t] = wo * stride - pad;
    ibase[t] = (size_t)ni * H * W * Cin;
  }

  const int KHW = KH * KW;
  int chunk = 0;
  for (int cell = 0; cell < KHW; cell++) {
    const int kh = cell / KW;
    const int kw = cell - kh * KW;
    for (int cb = 0; cb < Cin; cb += 32, chunk++) {
      const int buf = chunk & 1;
      const int k0 = cell * Cin + cb;
      // stage 32x32 weight chunk (2 subtiles), pre-swizzled into fragment order
      for (int i = threadIdx.x; i < 1024; i += 256) {
        int sub = i >> 9;
        int li = (i >> 4) & 31;
        int e  = i & 15;
        int mm = li & 15, lh = li >> 4;
        int ka = e + (lh << 4);
        int mg = m0 + sub * 16 + mm; if (mg >= Cout) mg = Cout - 1;
        As[buf][i] = wpk[(size_t)mg * Ktot + k0 + ka];
      }
      __syncthreads();   // single barrier per chunk (double-buffered stage)

      bf16x16 a0 = *(const bf16x16*)(&As[buf][0]   + (lane << 4));  // 2x ds_load_b128
      bf16x16 a1 = *(const bf16x16*)(&As[buf][512] + (lane << 4));
      __builtin_prefetch(wpk + (size_t)m0 * Ktot + ((k0 + 32 < Ktot) ? (k0 + 32) : 0), 0, 1);

      bf16x16 bfr[2];
#pragma unroll
      for (int t = 0; t < 2; t++) {
        int hh = hb[t] + kh, ww = wb[t] + kw;
        bf16x8 lo = {}, hp = {};
        if ((unsigned)hh < (unsigned)H && (unsigned)ww < (unsigned)W) {
          const bf16x8* bp = (const bf16x8*)(in + ibase[t] +
                              (size_t)(hh * W + ww) * Cin + cb + (hw2 << 4));
          lo = bp[0];   // global_load_b128
          hp = bp[1];   // global_load_b128
        }
        bfr[t] = __builtin_shufflevector(lo, hp,
            0,1,2,3,4,5,6,7,8,9,10,11,12,13,14,15);
      }
      acc[0][0] = __builtin_amdgcn_wmma_f32_16x16x32_bf16(false, a0, false, bfr[0],
                                                          (short)0, acc[0][0], false, false);
      acc[0][1] = __builtin_amdgcn_wmma_f32_16x16x32_bf16(false, a0, false, bfr[1],
                                                          (short)0, acc[0][1], false, false);
      acc[1][0] = __builtin_amdgcn_wmma_f32_16x16x32_bf16(false, a1, false, bfr[0],
                                                          (short)0, acc[1][0], false, false);
      acc[1][1] = __builtin_amdgcn_wmma_f32_16x16x32_bf16(false, a1, false, bfr[1],
                                                          (short)0, acc[1][1], false, false);
    }
  }

  // epilogue: BN + residual + ReLU; bf16 out is NHWC, f32 out is NCHW-strided
#pragma unroll
  for (int t = 0; t < 2; t++) {
    int ng = n0 + t * 16 + mrow;
    if (ng >= Ncol) continue;
    int ni = ng / HoWo; int rr = ng - ni * HoWo;
    size_t obase = ((size_t)ni * HoWo + rr) * Cout;
#pragma unroll
    for (int s = 0; s < 2; s++) {
#pragma unroll
      for (int j = 0; j < 8; j++) {
        int mg = m0 + s * 16 + j + (hw2 << 3);
        if (mg >= Cout) continue;
        float vv = acc[s][t][j];
        if (scale) vv = vv * scale[mg] + bias[mg];
        if (outf) {
          outf[(size_t)ni * obstride + ooff + (size_t)mg * HoWo + rr] = vv;
        } else {
          size_t oidx = obase + mg;
          if (resid) vv += (float)resid[oidx];
          if (relu && vv < 0.f) vv = 0.f;
          outb[oidx] = (__bf16)vv;
        }
      }
    }
  }
}

// ---------------- generic conv (conv1, Cin=3), NHWC, 16x16 tile/wave -------

__global__ void __launch_bounds__(256) conv_wmma_generic_kernel(
    const __bf16* __restrict__ in, const __bf16* __restrict__ wpk,
    const float* __restrict__ scale, const float* __restrict__ bias,
    __bf16* __restrict__ outb,
    int B, int Cin, int H, int W, int Cout, int Ho, int Wo,
    int KH, int KW, int stride, int pad, int relu)
{
  const int HoWo = Ho * Wo;
  const int Ncol = B * HoWo;
  const int Ktot = Cin * KH * KW;
  const int lane = threadIdx.x & 31;
  const int wave = threadIdx.x >> 5;
  const int m0 = blockIdx.y * 16;
  const int n0 = (blockIdx.x * 8 + wave) * 16;
  const int mrow = lane & 15;
  const int hw2 = lane >> 4;

  __shared__ __bf16 As[512];

  f32x8 acc = {0.f,0.f,0.f,0.f,0.f,0.f,0.f,0.f};

  int ncol = n0 + mrow; if (ncol >= Ncol) ncol = Ncol - 1;
  const int ni0 = ncol / HoWo;
  const int rc  = ncol - ni0 * HoWo;
  const int ho = rc / Wo;
  const int wo = rc - ho * Wo;
  const int hbase = ho * stride - pad;
  const int wbase = wo * stride - pad;
  const size_t ibase = (size_t)ni0 * H * W * Cin;

  for (int kb = 0; kb < Ktot; kb += 32) {
    for (int i = threadIdx.x; i < 512; i += 256) {
      int mm = i >> 5, kk = i & 31;
      int mg = m0 + mm; if (mg >= Cout) mg = Cout - 1;
      int kg = kb + kk;
      As[i] = (kg < Ktot) ? wpk[(size_t)mg * Ktot + kg] : (__bf16)0.0f;
    }
    __syncthreads();
    bf16x16 afr, bfr;
#pragma unroll
    for (int e = 0; e < 16; e++) {
      int ka = e + (hw2 << 4);
      afr[e] = As[(mrow << 5) + ka];
      int kg = kb + e + (hw2 << 4);      // B: lanes 0-15 K 0..15, lanes 16-31 K 16..31
      float bv = 0.f;
      if (kg < Ktot) {
        int cell = kg / Cin;
        int ci = kg - cell * Cin;
        int kh = cell / KW;
        int kw = cell - kh * KW;
        int hh = hbase + kh, ww = wbase + kw;
        if ((unsigned)hh < (unsigned)H && (unsigned)ww < (unsigned)W)
          bv = (float)in[ibase + (size_t)(hh * W + ww) * Cin + ci];
      }
      bfr[e] = (__bf16)bv;
    }
    acc = __builtin_amdgcn_wmma_f32_16x16x32_bf16(false, afr, false, bfr,
                                                  (short)0, acc, false, false);
    __syncthreads();
  }

#pragma unroll
  for (int j = 0; j < 8; j++) {
    int mg = m0 + j + (hw2 << 3);
    int ng = n0 + mrow;
    if (mg < Cout && ng < Ncol) {
      float vv = acc[j];
      if (scale) vv = vv * scale[mg] + bias[mg];
      int ni = ng / HoWo; int rr = ng - ni * HoWo;
      if (relu && vv < 0.f) vv = 0.f;
      outb[((size_t)ni * HoWo + rr) * Cout + mg] = (__bf16)vv;
    }
  }
}

// ------------------------------- pooling (NHWC) ----------------------------

__global__ void __launch_bounds__(256) maxpool_nhwc_kernel(
    const __bf16* __restrict__ in, __bf16* __restrict__ out, int N, int C, int H, int W) {
  int Ho = H / 2, Wo = W / 2;
  int total = N * Ho * Wo * C;
  int idx = blockIdx.x * blockDim.x + threadIdx.x;
  if (idx >= total) return;
  int c = idx % C; int r = idx / C;
  int wo = r % Wo; r /= Wo;
  int ho = r % Ho; int n = r / Ho;
  float m = -INFINITY;
  for (int dh = 0; dh < 3; dh++)
    for (int dw = 0; dw < 3; dw++) {
      int hh = ho * 2 - 1 + dh, ww = wo * 2 - 1 + dw;
      if (hh >= 0 && hh < H && ww >= 0 && ww < W) {
        float v = (float)in[(((size_t)n * H + hh) * W + ww) * C + c];
        if (v > m) m = v;
      }
    }
  out[idx] = (__bf16)m;
}

__global__ void avgpool_nhwc_kernel(const __bf16* __restrict__ in, float* __restrict__ out,
                                    int N, int C, int HW) {
  int idx = blockIdx.x * blockDim.x + threadIdx.x;
  if (idx >= N * C) return;
  int n = idx / C, c = idx % C;
  const __bf16* p = in + (size_t)n * HW * C + c;
  float s = 0.f;
  for (int i = 0; i < HW; i++) s += (float)p[(size_t)i * C];
  out[idx] = s / (float)HW;
}

// ------------------------------ small GEMM/FC ------------------------------

__global__ void fc_kernel(const float* __restrict__ in, const float* __restrict__ w,
                          const float* __restrict__ bias, float* __restrict__ out,
                          int M, int K, int N) {
  int idx = blockIdx.x * blockDim.x + threadIdx.x;
  if (idx >= M * N) return;
  int j = idx % N, i = idx / N;
  float a = bias ? bias[j] : 0.f;
  const float* ir = in + (size_t)i * K;
  const float* wr = w  + (size_t)j * K;
  for (int k = 0; k < K; k++) a += ir[k] * wr[k];
  out[idx] = a;
}

__global__ void concat_build_kernel(const float* __restrict__ pfeat,
                                    const float* __restrict__ feat,
                                    float* __restrict__ out) {
  int idx = blockIdx.x * blockDim.x + threadIdx.x;
  if (idx >= 8 * 2560) return;
  int b = idx / 2560, j = idx % 2560;
  out[idx] = (j < 2048) ? pfeat[b * 2048 + j] : feat[b * 512 + (j - 2048)];
}

// --------------------------------- NMS -------------------------------------

__global__ void nms_kernel(const float* __restrict__ score, const int* __restrict__ anchors,
                           int* __restrict__ selbox,
                           float* __restrict__ out_idx, float* __restrict__ out_prob) {
  int b = blockIdx.x;
  __shared__ float s[N_ANCH];
  __shared__ float rmax[256];
  __shared__ int   rarg[256];
  __shared__ int   pick;
  const float* sb = score + b * N_ANCH;
  for (int i = threadIdx.x; i < N_ANCH; i += 256) s[i] = sb[i];
  __syncthreads();
  for (int t = 0; t < 4; t++) {
    float m = -INFINITY; int mi = 0;
    for (int i = threadIdx.x; i < N_ANCH; i += 256)
      if (s[i] > m) { m = s[i]; mi = i; }
    rmax[threadIdx.x] = m; rarg[threadIdx.x] = mi;
    __syncthreads();
    if (threadIdx.x == 0) {
      float bm = rmax[0]; int ba = rarg[0];
      for (int i = 1; i < 256; i++)
        if (rmax[i] > bm) { bm = rmax[i]; ba = rarg[i]; }
      pick = ba;
      out_idx[b * 4 + t]  = (float)ba;
      out_prob[b * 4 + t] = sb[ba];
      for (int k = 0; k < 4; k++) selbox[(b * 4 + t) * 4 + k] = anchors[ba * 4 + k];
    }
    __syncthreads();
    int pi = pick;
    float py0 = (float)anchors[pi * 4 + 0], px0 = (float)anchors[pi * 4 + 1];
    float py1 = (float)anchors[pi * 4 + 2], px1 = (float)anchors[pi * 4 + 3];
    float parea = (py1 - py0) * (px1 - px0);
    for (int i = threadIdx.x; i < N_ANCH; i += 256) {
      float y0 = (float)anchors[i * 4 + 0], x0 = (float)anchors[i * 4 + 1];
      float y1 = (float)anchors[i * 4 + 2], x1 = (float)anchors[i * 4 + 3];
      float iy0 = fmaxf(y0, py0), ix0 = fmaxf(x0, px0);
      float iy1 = fminf(y1, py1), ix1 = fminf(x1, px1);
      float inter = fmaxf(iy1 - iy0, 0.f) * fmaxf(ix1 - ix0, 0.f);
      float area  = (y1 - y0) * (x1 - x0);
      float iou = inter / (area + parea - inter);
      if (iou > 0.5f) s[i] = -INFINITY;
    }
    __syncthreads();
    if (threadIdx.x == 0) s[pi] = -INFINITY;
    __syncthreads();
  }
}

// ------------------------------ crop + resize ------------------------------
// Bilinear crop from (implicitly zero-padded by 224) 448x448 NCHW f32 images,
// writing NHWC bf16 part images [32][224][224][3].

__global__ void __launch_bounds__(256) crop_resize_kernel(
    const float* __restrict__ x, const int* __restrict__ selbox, __bf16* __restrict__ out) {
  int idx = blockIdx.x * blockDim.x + threadIdx.x;
  const int total = 32 * 224 * 224 * 3;
  if (idx >= total) return;
  int c = idx % 3;   int r = idx / 3;
  int j = r % 224;   r /= 224;
  int i = r % 224;   int p = r / 224;
  int b = p >> 2;
  const int* bx = selbox + p * 4;
  float y0 = (float)bx[0], x0 = (float)bx[1], y1 = (float)bx[2], x1 = (float)bx[3];
  float ty = (float)i / 223.f, tx = (float)j / 223.f;
  float ys = y0 + ty * (y1 - 1.f - y0);
  float xs = x0 + tx * (x1 - 1.f - x0);
  float yf = floorf(ys), xf = floorf(xs);
  float wy = ys - yf,    wx = xs - xf;
  int y0i = (int)yf, x0i = (int)xf;
  int y1i = (y0i + 1 < 895) ? y0i + 1 : 895;
  int x1i = (x0i + 1 < 895) ? x0i + 1 : 895;
  const float* img = x + ((size_t)b * 3 + c) * 448 * 448;
  auto sample = [&](int yy, int xx) -> float {
    int yo = yy - 224, xo = xx - 224;
    if (yo < 0 || yo >= 448 || xo < 0 || xo >= 448) return 0.f;
    return img[yo * 448 + xo];
  };
  float v = sample(y0i, x0i) * (1.f - wy) * (1.f - wx)
          + sample(y0i, x1i) * (1.f - wy) * wx
          + sample(y1i, x0i) * wy * (1.f - wx)
          + sample(y1i, x1i) * wy * wx;
  out[idx] = (__bf16)v;
}

// -------------------------------- BiLSTM -----------------------------------

__global__ void lstm_kernel(const float* __restrict__ pfeat, const float* __restrict__ feat,
                            const float* __restrict__ wihf, const float* __restrict__ whhf,
                            const float* __restrict__ wihb, const float* __restrict__ whhb,
                            float* __restrict__ out /* [8][5][18] */) {
  int b = blockIdx.x;
  __shared__ float h[9], c[9], g[36];
  __shared__ float xt[512];
  int tid = threadIdx.x;  // blockDim = 64
  for (int dir = 0; dir < 2; dir++) {
    const float* wih = dir ? wihb : wihf;
    const float* whh = dir ? whhb : whhf;
    if (tid < 9) { h[tid] = 0.f; c[tid] = 0.f; }
    __syncthreads();
    for (int t = 0; t < 5; t++) {
      int step = dir ? (4 - t) : t;
      const float* src = (step < 4) ? (pfeat + (size_t)(b * 4 + step) * 512)
                                    : (feat + (size_t)b * 512);
      for (int i = tid; i < 512; i += 64) xt[i] = src[i];
      __syncthreads();
      if (tid < 36) {
        float a = 0.f;
        const float* wr = wih + (size_t)tid * 512;
        for (int k = 0; k < 512; k++) a += wr[k] * xt[k];
        const float* hr = whh + (size_t)tid * 9;
        for (int k = 0; k < 9; k++) a += hr[k] * h[k];
        g[tid] = a;
      }
      __syncthreads();
      if (tid < 9) {
        float ig = 1.f / (1.f + expf(-g[tid]));
        float fg = 1.f / (1.f + expf(-g[9 + tid]));
        float gg = tanhf(g[18 + tid]);
        float og = 1.f / (1.f + expf(-g[27 + tid]));
        float cn = fg * c[tid] + ig * gg;
        float hn = og * tanhf(cn);
        c[tid] = cn; h[tid] = hn;
        out[(b * 5 + step) * 18 + dir * 9 + tid] = hn;
      }
      __syncthreads();
    }
    __syncthreads();
  }
}

// =============================== host side =================================

extern "C" void kernel_launch(void* const* d_in, const int* in_sizes, int n_in,
                              void* d_out, int out_size, void* d_ws, size_t ws_size,
                              hipStream_t stream) {
  (void)in_sizes; (void)n_in; (void)out_size; (void)ws_size;

  // ---- input cursor (setup_inputs() insertion order) ----
  int cur = 0;
  auto nextf = [&]() { return (const float*)d_in[cur++]; };

  const float* xin = nextf();                                  // x [8,3,448,448]
  const float* rconv1 = nextf();                               // resnet.conv1
  const float* rbn1[4]; for (int k = 0; k < 4; k++) rbn1[k] = nextf();
  const int cinA[8]  = {64, 64, 64, 128, 128, 256, 256, 512};
  const int coutA[8] = {64, 64, 128, 128, 256, 256, 512, 512};
  const int bst[8]   = {1, 1, 2, 1, 2, 1, 2, 1};
  const bool hasdown[8] = {false, false, true, false, true, false, true, false};
  const float *bc1[8], *bb1[8][4], *bc2[8], *bb2[8][4], *bdw[8], *bdb[8][4];
  for (int i = 0; i < 8; i++) {
    bc1[i] = nextf(); for (int k = 0; k < 4; k++) bb1[i][k] = nextf();
    bc2[i] = nextf(); for (int k = 0; k < 4; k++) bb2[i][k] = nextf();
    bdw[i] = nullptr;
    if (hasdown[i]) { bdw[i] = nextf(); for (int k = 0; k < 4; k++) bdb[i][k] = nextf(); }
  }
  const float* fcw = nextf(); const float* fcb = nextf();
  const float* pdown1 = nextf(); const float* pdown2 = nextf(); const float* pdown3 = nextf();
  const float* ptidy1 = nextf(); const float* ptidy2 = nextf(); const float* ptidy3 = nextf();
  const float* wihf = nextf(); const float* whhf = nextf();
  const float* wihb = nextf(); const float* whhb = nextf();
  const float* concw = nextf(); const float* concb = nextf();
  const float* partw = nextf(); const float* partbias = nextf();
  const int* anchors = (const int*)d_in[cur++];

  float* dout = (float*)d_out;

  // ---- workspace bump allocator ----
  size_t off = 0;
  auto alloc = [&](size_t bytes) -> void* {
    size_t a = (off + 255) & ~(size_t)255;
    off = a + bytes;
    return (void*)((char*)d_ws + a);
  };
  auto repack = [&](const float* src, int Cout, int Cin, int K) -> __bf16* {
    int n = Cout * Cin * K * K;
    __bf16* dst = (__bf16*)alloc((size_t)n * 2);
    repack_w_kernel<<<(n + 255) / 256, 256, 0, stream>>>(src, dst, Cout, Cin, K, K);
    return dst;
  };
  auto fold = [&](const float* const bn[4], int C, float** sc, float** bi) {
    *sc = (float*)alloc((size_t)C * 4);
    *bi = (float*)alloc((size_t)C * 4);
    foldbn_kernel<<<(C + 63) / 64, 64, 0, stream>>>(bn[0], bn[1], bn[2], bn[3], *sc, *bi, C);
  };

  // ---- repack weights to bf16 [Cout][KH*KW*Cin], fold BN ----
  __bf16* wc1 = repack(rconv1, 64, 3, 7);
  float *sc1, *bi1; fold(rbn1, 64, &sc1, &bi1);
  __bf16 *wbc1[8], *wbc2[8], *wbd[8];
  float *s1[8], *o1[8], *s2[8], *o2[8], *sdn[8], *odn[8];
  for (int i = 0; i < 8; i++) {
    wbc1[i] = repack(bc1[i], coutA[i], cinA[i], 3);
    wbc2[i] = repack(bc2[i], coutA[i], coutA[i], 3);
    fold(bb1[i], coutA[i], &s1[i], &o1[i]);
    fold(bb2[i], coutA[i], &s2[i], &o2[i]);
    wbd[i] = nullptr;
    if (hasdown[i]) {
      wbd[i] = repack(bdw[i], coutA[i], cinA[i], 1);
      fold(bdb[i], coutA[i], &sdn[i], &odn[i]);
    }
  }
  __bf16* wpd1 = repack(pdown1, 128, 512, 3);
  __bf16* wpd2 = repack(pdown2, 128, 128, 3);
  __bf16* wpd3 = repack(pdown3, 128, 128, 3);
  __bf16* wpt1 = repack(ptidy1, 6, 128, 1);
  __bf16* wpt2 = repack(ptidy2, 6, 128, 1);
  __bf16* wpt3 = repack(ptidy3, 9, 128, 1);

  // ---- activation buffers (all NHWC bf16) ----
  __bf16* XB   = (__bf16*)alloc((size_t)4816896 * 2);   // x bf16 / part imgs
  __bf16* BIG0 = (__bf16*)alloc((size_t)25690112 * 2);  // conv1 output
  __bf16* PB[5];
  for (int i = 0; i < 5; i++) PB[i] = (__bf16*)alloc((size_t)6422528 * 2);
  __bf16* RPNF = (__bf16*)alloc((size_t)802816 * 2);    // layer4 out (both passes)
  float* FEAT  = (float*)alloc((size_t)8 * 512 * 4);
  float* PFEAT = (float*)alloc((size_t)32 * 512 * 4);
  float* RPNSC = (float*)alloc((size_t)8 * N_ANCH * 4);
  float* CONC  = (float*)alloc((size_t)8 * 2560 * 4);
  int* SELBOX  = (int*)alloc((size_t)128 * 4);

  // ---- conv launchers ----
  auto conv = [&](const __bf16* in, const __bf16* w, const float* sc, const float* bi,
                  const __bf16* resid, __bf16* outb, float* outf,
                  int N, int Cin, int H, int Cout, int KH, int st, int pad, int relu,
                  int obstride, int ooff) {
    int Ho = (H + 2 * pad - KH) / st + 1;
    int Ncol = N * Ho * Ho;
    dim3 grid((Ncol + 255) / 256, (Cout + 31) / 32, 1);
    conv_wmma_fast_kernel<<<grid, dim3(256, 1, 1), 0, stream>>>(
        in, w, sc, bi, resid, outb, outf,
        N, Cin, H, H, Cout, Ho, Ho, KH, KH, st, pad, relu, obstride, ooff);
  };
  auto conv_g = [&](const __bf16* in, const __bf16* w, const float* sc, const float* bi,
                    __bf16* outb, int N, int Cin, int H, int Cout, int KH, int st,
                    int pad, int relu) {
    int Ho = (H + 2 * pad - KH) / st + 1;
    int Ncol = N * Ho * Ho;
    dim3 grid((Ncol + 127) / 128, (Cout + 15) / 16, 1);
    conv_wmma_generic_kernel<<<grid, dim3(256, 1, 1), 0, stream>>>(
        in, w, sc, bi, outb, N, Cin, H, H, Cout, Ho, Ho, KH, KH, st, pad, relu);
  };

  // ---- full ResNet-18 pass (conv1+maxpool+8 blocks), NHWC ----
  auto resnet_pass = [&](const __bf16* inp, int N, int Himg, __bf16* finalout) {
    conv_g(inp, wc1, sc1, bi1, BIG0, N, 3, Himg, 64, 7, 2, 3, 1);
    int Hc = Himg / 2;
    {
      int Ho = Hc / 2;
      int total = N * 64 * Ho * Ho;
      maxpool_nhwc_kernel<<<(total + 255) / 256, 256, 0, stream>>>(BIG0, PB[4], N, 64, Hc, Hc);
    }
    const __bf16* curp = PB[4];
    int curi = 4;
    int Hcur = Hc / 2;
    for (int i = 0; i < 8; i++) {
      int st = bst[i];
      int Ho = (st == 2) ? Hcur / 2 : Hcur;
      int used[3]; int u = 0;
      auto pick = [&]() -> int {
        for (int k = 0; k < 5; k++) {
          if (k == curi) continue;
          bool ok = true;
          for (int q = 0; q < u; q++) if (used[q] == k) ok = false;
          if (ok) { used[u++] = k; return k; }
        }
        return 0;
      };
      int ti = pick();
      __bf16* t = PB[ti];
      __bf16* outp; int oi = -1;
      if (i == 7) outp = finalout;
      else { oi = pick(); outp = PB[oi]; }
      conv(curp, wbc1[i], s1[i], o1[i], nullptr, t, nullptr,
           N, cinA[i], Hcur, coutA[i], 3, st, 1, 1, 0, 0);
      const __bf16* resid = curp;
      if (hasdown[i]) {
        int si = pick();
        conv(curp, wbd[i], sdn[i], odn[i], nullptr, PB[si], nullptr,
             N, cinA[i], Hcur, coutA[i], 1, st, 0, 0, 0, 0);
        resid = PB[si];
      }
      conv(t, wbc2[i], s2[i], o2[i], resid, outp, nullptr,
           N, coutA[i], Ho, coutA[i], 3, 1, 1, 1, 0, 0);
      curp = outp; curi = oi; Hcur = Ho;
    }
  };

  // ================= pipeline =================
  // 1) x -> NHWC bf16
  nchw_to_nhwc_kernel<<<(4816896 + 255) / 256, 256, 0, stream>>>(xin, XB, 8, 3, 448, 448);

  // 2) main ResNet pass (N=8, 448) -> RPNF = rpn_feature [8,14,14,512]
  resnet_pass(XB, 8, 448, RPNF);

  // 3) feature = mean(h) ; raw_logits
  avgpool_nhwc_kernel<<<(8 * 512 + 255) / 256, 256, 0, stream>>>(RPNF, FEAT, 8, 512, 196);
  fc_kernel<<<1, 256, 0, stream>>>(FEAT, fcw, fcb, dout + 0, 8, 512, 9);

  // 4) proposal net -> rpn_score [8,2793] (tidy outputs stored NCHW-strided f32)
  conv(RPNF, wpd1, nullptr, nullptr, nullptr, PB[0], nullptr, 8, 512, 14, 128, 3, 1, 1, 1, 0, 0);
  conv(PB[0], wpd2, nullptr, nullptr, nullptr, PB[1], nullptr, 8, 128, 14, 128, 3, 1, 1, 1, 0, 0);
  conv(PB[1], wpd3, nullptr, nullptr, nullptr, PB[2], nullptr, 8, 128, 14, 128, 3, 2, 1, 1, 0, 0);
  conv(PB[0], wpt1, nullptr, nullptr, nullptr, nullptr, RPNSC, 8, 128, 14, 6, 1, 1, 0, 0, N_ANCH, 0);
  conv(PB[1], wpt2, nullptr, nullptr, nullptr, nullptr, RPNSC, 8, 128, 14, 6, 1, 1, 0, 0, N_ANCH, 1176);
  conv(PB[2], wpt3, nullptr, nullptr, nullptr, nullptr, RPNSC, 8, 128, 7, 9, 1, 1, 0, 0, N_ANCH, 2352);

  // 5) per-batch hard NMS -> top-4 indices/probs/boxes
  nms_kernel<<<8, 256, 0, stream>>>(RPNSC, anchors, SELBOX, dout + 432, dout + 464);

  // 6) crop+resize 32 part images (NHWC bf16), reusing XB
  {
    int total = 32 * 224 * 224 * 3;
    crop_resize_kernel<<<(total + 255) / 256, 256, 0, stream>>>(xin, SELBOX, XB);
  }

  // 7) part ResNet pass (N=32, 224) -> RPNF reused as [32,7,7,512]
  resnet_pass(XB, 32, 224, RPNF);
  avgpool_nhwc_kernel<<<(32 * 512 + 255) / 256, 256, 0, stream>>>(RPNF, PFEAT, 32, 512, 49);

  // 8) part logits [32,9] -> d_out[144:432]
  fc_kernel<<<(32 * 9 + 255) / 256, 256, 0, stream>>>(PFEAT, partw, partbias, dout + 144, 32, 512, 9);

  // 9) concat logits
  concat_build_kernel<<<(8 * 2560 + 255) / 256, 256, 0, stream>>>(PFEAT, FEAT, CONC);
  fc_kernel<<<1, 256, 0, stream>>>(CONC, concw, concb, dout + 72, 8, 2560, 9);

  // 10) BiLSTM -> d_out[496:1216]
  lstm_kernel<<<8, 64, 0, stream>>>(PFEAT, FEAT, wihf, whhf, wihb, whhb, dout + 496);
}
